// Attention_50843822850512
// MI455X (gfx1250) — compile-verified, matrix operand order checked by reading, not run on hardware
//
#include <hip/hip_runtime.h>
#include <math.h>

typedef __attribute__((ext_vector_type(16))) _Float16 v16h;
typedef __attribute__((ext_vector_type(8)))  _Float16 v8h;
typedef __attribute__((ext_vector_type(8)))  float    v8f;

#define B_    2
#define S_    1024
#define D_    4096
#define H_    32
#define KVH_  8
#define HD_   128
#define MAXC  2048
#define REP_  (H_ / KVH_)
#define THETA 1000000.0f

// ---------------------------------------------------------------------------
// WMMA helpers
// ---------------------------------------------------------------------------
union AFrag { v16h v; struct { v8h lo, hi; } s; };

// Load a 16x32 f16 fragment (A or B operand) from a row-major [rows, ldm]
// matrix (global or LDS). Row within tile = lane&15; lanes 16..31 take K
// offset +8. Halves 0..7 = K{0..7}, halves 8..15 = K{16..23}.
__device__ __forceinline__ v16h load_frag(const _Float16* __restrict__ base,
                                          int row0, int ldm, int kbase, int lane) {
  int r  = lane & 15;
  int lh = lane >> 4;
  const _Float16* p = base + (size_t)(row0 + r) * ldm + kbase + lh * 8;
  AFrag f;
  f.s.lo = *(const v8h*)(p);
  f.s.hi = *(const v8h*)(p + 16);
  return f.v;
}

__device__ __forceinline__ v8f wmma_f16(v16h a, v16h b, v8f c) {
  return __builtin_amdgcn_wmma_f32_16x16x32_f16(false, a, false, b,
                                                (short)0, c, false, false);
}

// CDNA5 async global->LDS copy (16B per lane), tracked by ASYNCcnt.
__device__ __forceinline__ void async_ld_b128(unsigned lds_addr, uint64_t gaddr) {
  asm volatile("global_load_async_to_lds_b128 %0, %1, off"
               :: "v"(lds_addr), "v"(gaddr) : "memory");
}
__device__ __forceinline__ void wait_async_le4() {
  asm volatile("s_wait_asynccnt 0x4" ::: "memory");
}
__device__ __forceinline__ void wait_async_0() {
  asm volatile("s_wait_asynccnt 0x0" ::: "memory");
}

// ---------------------------------------------------------------------------
// Conversion / transpose kernels
// ---------------------------------------------------------------------------
__global__ void cvt_f32_to_f16(const float* __restrict__ in,
                               _Float16* __restrict__ out, int n) {
  int i = blockIdx.x * blockDim.x + threadIdx.x;
  if (i < n) out[i] = (_Float16)in[i];
}

// Wt[n*K + k] = (f16) W[k*N + n]   (W row-major [K,N] -> Wt row-major [N,K])
__global__ void transpose_to_f16(const float* __restrict__ W,
                                 _Float16* __restrict__ Wt, int K, int N) {
  int idx = blockIdx.x * blockDim.x + threadIdx.x;
  if (idx >= K * N) return;
  int k = idx % K;
  int n = idx / K;
  Wt[(size_t)n * K + k] = (_Float16)W[(size_t)k * N + n];
}

// ---------------------------------------------------------------------------
// WMMA GEMM: C[M,N] = A[M,K] * Bt[N,K]^T + bias[N]
// Block = 256 threads = 8 waves (4 M-waves x 2 N-waves), block tile 128x128.
// Per K-step (32): 128x32 A/B panels async-staged to LDS (double-buffered,
// ASYNCcnt), each wave then does 8 WMMAs on a 32x64 register tile.
// ---------------------------------------------------------------------------
#define KSTEP 32
#define TILE_HALF (128 * KSTEP)            // halves per panel
#define TILE_BYTES (TILE_HALF * 2)         // 8 KB

__global__ void gemm_f16_wmma(const _Float16* __restrict__ A,
                              const _Float16* __restrict__ Bt,
                              const float* __restrict__ bias,
                              float* __restrict__ C,
                              int M, int N, int K) {
  __shared__ _Float16 Atile[2][TILE_HALF];
  __shared__ _Float16 Btile[2][TILE_HALF];

  const int t    = threadIdx.x;
  const int lane = t & 31;
  const int wave = t >> 5;
  const int wm   = wave >> 1;     // 0..3
  const int wn   = wave & 1;      // 0..1
  const int r    = lane & 15;
  const int lh   = lane >> 4;

  const int nblk = blockIdx.x * 128;
  const int mblk = blockIdx.y * 128;

  const unsigned abase = (unsigned)(uintptr_t)&Atile[0][0];
  const unsigned bbase = (unsigned)(uintptr_t)&Btile[0][0];

  // Issue the 4 async b128 loads (2 for A panel, 2 for B panel) for one stage.
  auto stage = [&](int buf, int k) {
#pragma unroll
    for (int j = 0; j < 2; ++j) {
      int idx = t * 2 + j;            // 0..511
      int off = idx * 8;              // half index in panel
      int row = off >> 5;             // /32
      int col = off & 31;
      uint64_t ga = (uint64_t)(uintptr_t)(A + (size_t)(mblk + row) * K + k + col);
      async_ld_b128(abase + (unsigned)(buf * TILE_BYTES + off * 2), ga);
      uint64_t gb = (uint64_t)(uintptr_t)(Bt + (size_t)(nblk + row) * K + k + col);
      async_ld_b128(bbase + (unsigned)(buf * TILE_BYTES + off * 2), gb);
    }
  };

  v8f acc[2][4];
#pragma unroll
  for (int mi = 0; mi < 2; ++mi)
#pragma unroll
    for (int ni = 0; ni < 4; ++ni) acc[mi][ni] = (v8f){};

  stage(0, 0);
  for (int kb = 0; kb < K; kb += KSTEP) {
    int buf = (kb >> 5) & 1;
    bool more = (kb + KSTEP) < K;
    if (more) {
      stage(buf ^ 1, kb + KSTEP);
      wait_async_le4();   // oldest 4 (this buf) done; async loads retire in order
    } else {
      wait_async_0();
    }
    __syncthreads();

    v16h af[2];
#pragma unroll
    for (int mi = 0; mi < 2; ++mi)
      af[mi] = load_frag(&Atile[buf][0], wm * 32 + mi * 16, KSTEP, 0, lane);
    v16h bf[4];
#pragma unroll
    for (int ni = 0; ni < 4; ++ni)
      bf[ni] = load_frag(&Btile[buf][0], wn * 64 + ni * 16, KSTEP, 0, lane);

#pragma unroll
    for (int mi = 0; mi < 2; ++mi)
#pragma unroll
      for (int ni = 0; ni < 4; ++ni)
        acc[mi][ni] = wmma_f16(af[mi], bf[ni], acc[mi][ni]);

    __syncthreads();
  }

#pragma unroll
  for (int ni = 0; ni < 4; ++ni) {
    int col = nblk + wn * 64 + ni * 16 + r;
    float bv = bias ? bias[col] : 0.0f;
#pragma unroll
    for (int mi = 0; mi < 2; ++mi) {
#pragma unroll
      for (int i = 0; i < 8; ++i) {
        int row = mblk + wm * 32 + mi * 16 + i + 8 * lh;
        C[(size_t)row * N + col] = acc[mi][ni][i] + bv;
      }
    }
  }
}

// ---------------------------------------------------------------------------
// RoPE + layout kernels
// ---------------------------------------------------------------------------
__device__ __forceinline__ float rope_val(const float* __restrict__ row,
                                          int d, int pos) {
  int   half = HD_ / 2;
  float t  = row[d];
  float tp = (d < half) ? -row[d + half] : row[d - half];
  int   j  = (d < half) ? d : d - half;
  float inv = powf(THETA, -((float)(2 * j) / (float)HD_));
  float ang = (float)pos * inv;
  return t * cosf(ang) + tp * sinf(ang);
}

// qf [B*S, H*HD] f32 -> Qh [B,H,S,HD] f16 with RoPE
__global__ void rope_q_kernel(const float* __restrict__ qf,
                              const int* __restrict__ sidx,
                              _Float16* __restrict__ Qh) {
  int idx = blockIdx.x * blockDim.x + threadIdx.x;
  if (idx >= B_ * S_ * H_ * HD_) return;
  int d = idx % HD_;
  int h = (idx / HD_) % H_;
  int s = (idx / (HD_ * H_)) % S_;
  int b = idx / (HD_ * H_ * S_);
  int pos = sidx[s];
  const float* row = qf + ((size_t)(b * S_ + s)) * (H_ * HD_) + (size_t)h * HD_;
  float o = rope_val(row, d, pos);
  Qh[(((size_t)b * H_ + h) * S_ + s) * HD_ + d] = (_Float16)o;
}

// kf [B*S, KVH*HD] f32 -> Kh [B,KVH,MAX,HD] f16 with RoPE, scattered at pos
__global__ void rope_k_kernel(const float* __restrict__ kf,
                              const int* __restrict__ sidx,
                              _Float16* __restrict__ Kh) {
  int idx = blockIdx.x * blockDim.x + threadIdx.x;
  if (idx >= B_ * S_ * KVH_ * HD_) return;
  int d = idx % HD_;
  int h = (idx / HD_) % KVH_;
  int s = (idx / (HD_ * KVH_)) % S_;
  int b = idx / (HD_ * KVH_ * S_);
  int pos = sidx[s];
  const float* row = kf + ((size_t)(b * S_ + s)) * (KVH_ * HD_) + (size_t)h * HD_;
  float o = rope_val(row, d, pos);
  Kh[(((size_t)b * KVH_ + h) * MAXC + pos) * HD_ + d] = (_Float16)o;
}

// vf [B*S, KVH*HD] f32 -> Vt [B,KVH,HD,MAX] f16 (transposed for PV B-frag)
__global__ void v_scatter_kernel(const float* __restrict__ vf,
                                 const int* __restrict__ sidx,
                                 _Float16* __restrict__ Vt) {
  int idx = blockIdx.x * blockDim.x + threadIdx.x;
  if (idx >= B_ * S_ * KVH_ * HD_) return;
  int d = idx % HD_;
  int h = (idx / HD_) % KVH_;
  int s = (idx / (HD_ * KVH_)) % S_;
  int b = idx / (HD_ * KVH_ * S_);
  int pos = sidx[s];
  float v = vf[((size_t)(b * S_ + s)) * (KVH_ * HD_) + (size_t)h * HD_ + d];
  Vt[(((size_t)b * KVH_ + h) * HD_ + d) * (size_t)MAXC + pos] = (_Float16)v;
}

// ---------------------------------------------------------------------------
// Attention: one block (256 threads = 8 waves) per (b, h, 16-query tile).
// Phase 1: QK^T via WMMA into LDS (masked + scaled).
// Phase 2: exact softmax in LDS (f32).
// Phase 3: P*V via WMMA, each wave owns a 16-wide HD slice.
// ---------------------------------------------------------------------------
__global__ void attn_kernel(const _Float16* __restrict__ Qh,
                            const _Float16* __restrict__ Kh,
                            const _Float16* __restrict__ Vt,
                            _Float16* __restrict__ Oh /* [B*S, H*HD] */) {
  extern __shared__ char smem[];
  float* sc   = (float*)smem;          // [16][MAXC]
  float* red  = sc + 16 * MAXC;        // [16][16]
  float* rowm = red + 256;             // [16]
  float* rows = rowm + 16;             // [16]

  int qt = blockIdx.x;
  int h  = blockIdx.y;
  int b  = blockIdx.z;
  int kvh = h / REP_;

  int tid  = threadIdx.x;
  int lane = tid & 31;
  int wave = tid >> 5;
  int r    = lane & 15;
  int lh   = lane >> 4;

  const _Float16* Qbase = Qh + (((size_t)b * H_ + h) * S_ + (size_t)qt * 16) * HD_;
  const _Float16* Kbase = Kh + ((size_t)b * KVH_ + kvh) * (size_t)MAXC * HD_;
  const _Float16* Vbase = Vt + ((size_t)b * KVH_ + kvh) * (size_t)HD_ * MAXC;

  // Preload the 4 Q fragments (d = 0,32,64,96) — reused for all k-tiles.
  v16h aq[4];
#pragma unroll
  for (int t = 0; t < 4; ++t) aq[t] = load_frag(Qbase, 0, HD_, t * 32, lane);

  const float scale = 0.08838834764831845f;  // 1/sqrt(HD)

  // ---- Phase 1: scores ----
  for (int kt = wave; kt < MAXC / 16; kt += 8) {
    v8f acc = {};
#pragma unroll
    for (int t = 0; t < 4; ++t) {
      v16h bf = load_frag(Kbase, kt * 16, HD_, t * 32, lane);
      acc = wmma_f16(aq[t], bf, acc);
    }
    int kcol = kt * 16 + r;
#pragma unroll
    for (int i = 0; i < 8; ++i) {
      int row  = i + 8 * lh;
      int qpos = qt * 16 + row;
      sc[row * MAXC + kcol] = (kcol <= qpos) ? acc[i] * scale : -1e9f;
    }
  }
  __syncthreads();

  // ---- Phase 2: softmax (16 threads per row) ----
  int row = tid >> 4;
  int sub = tid & 15;
  float m = -1e30f;
  for (int c = sub; c < MAXC; c += 16) m = fmaxf(m, sc[row * MAXC + c]);
  red[row * 16 + sub] = m;
  __syncthreads();
  if (sub == 0) {
    float mm = red[row * 16];
#pragma unroll
    for (int i = 1; i < 16; ++i) mm = fmaxf(mm, red[row * 16 + i]);
    rowm[row] = mm;
  }
  __syncthreads();
  float mm = rowm[row];
  float ssum = 0.0f;
  for (int c = sub; c < MAXC; c += 16) {
    float e = __expf(sc[row * MAXC + c] - mm);
    sc[row * MAXC + c] = e;
    ssum += e;
  }
  red[row * 16 + sub] = ssum;
  __syncthreads();
  if (sub == 0) {
    float tt = 0.0f;
#pragma unroll
    for (int i = 0; i < 16; ++i) tt += red[row * 16 + i];
    rows[row] = tt;
  }
  __syncthreads();
  float inv = 1.0f / rows[row];
  for (int c = sub; c < MAXC; c += 16) sc[row * MAXC + c] *= inv;
  __syncthreads();

  // ---- Phase 3: P*V, wave owns HD slice [n0, n0+16) ----
  int n0 = wave * 16;
  v8f acc = {};
  for (int k = 0; k < MAXC; k += 32) {
    union { v16h v; _Float16 hh[16]; } af;
    const float* prow = sc + r * MAXC + k + lh * 8;
#pragma unroll
    for (int j = 0; j < 8; ++j) {
      af.hh[j]     = (_Float16)prow[j];
      af.hh[8 + j] = (_Float16)prow[16 + j];
    }
    v16h bf = load_frag(Vbase, n0, MAXC, k, lane);
    acc = wmma_f16(af.v, bf, acc);
  }
#pragma unroll
  for (int i = 0; i < 8; ++i) {
    int rr = i + 8 * lh;
    size_t orow = (size_t)(b * S_ + qt * 16 + rr);
    Oh[orow * (H_ * HD_) + (size_t)h * HD_ + n0 + r] = (_Float16)acc[i];
  }
}

// ---------------------------------------------------------------------------
// Host launcher
// ---------------------------------------------------------------------------
extern "C" void kernel_launch(void* const* d_in, const int* in_sizes, int n_in,
                              void* d_out, int out_size, void* d_ws, size_t ws_size,
                              hipStream_t stream) {
  (void)in_sizes; (void)n_in; (void)out_size; (void)ws_size;

  const float* x    = (const float*)d_in[0];
  const int*   sidx = (const int*)d_in[1];
  // d_in[2] = cache (zeros, unused), d_in[3] = mask (causal, computed inline)
  const float* Wq = (const float*)d_in[4];
  const float* bq = (const float*)d_in[5];
  const float* Wk = (const float*)d_in[6];
  const float* bk = (const float*)d_in[7];
  const float* Wv = (const float*)d_in[8];
  const float* bv = (const float*)d_in[9];
  const float* Wo = (const float*)d_in[10];
  float* out = (float*)d_out;

  const int M   = B_ * S_;          // 2048
  const int NQ  = H_ * HD_;         // 4096
  const int NKV = KVH_ * HD_;       // 1024

  // -------- workspace carve (bytes, 256-aligned) --------
  char* ws = (char*)d_ws;
  size_t off = 0;
  auto carve = [&](size_t bytes) {
    char* p = ws + off;
    off += (bytes + 255) & ~(size_t)255;
    return p;
  };
  _Float16* xh   = (_Float16*)carve((size_t)M * D_ * 2);
  _Float16* WqT  = (_Float16*)carve((size_t)NQ * D_ * 2);
  _Float16* WkT  = (_Float16*)carve((size_t)NKV * D_ * 2);
  _Float16* WvT  = (_Float16*)carve((size_t)NKV * D_ * 2);
  _Float16* WoT  = (_Float16*)carve((size_t)D_ * NQ * 2);
  float*    qf   = (float*)carve((size_t)M * NQ * 4);
  float*    kf   = (float*)carve((size_t)M * NKV * 4);
  float*    vf   = (float*)carve((size_t)M * NKV * 4);
  _Float16* Qh   = (_Float16*)carve((size_t)B_ * H_ * S_ * HD_ * 2);
  _Float16* Kh   = (_Float16*)carve((size_t)B_ * KVH_ * MAXC * HD_ * 2);
  _Float16* Vt   = (_Float16*)carve((size_t)B_ * KVH_ * HD_ * MAXC * 2);
  _Float16* Oh   = (_Float16*)carve((size_t)M * NQ * 2);

  const int TPB = 256;

  // 1) convert x to f16
  {
    int n = M * D_;
    cvt_f32_to_f16<<<(n + TPB - 1) / TPB, TPB, 0, stream>>>(x, xh, n);
  }
  // 2) transpose+convert weights
  {
    int n = D_ * NQ;
    transpose_to_f16<<<(n + TPB - 1) / TPB, TPB, 0, stream>>>(Wq, WqT, D_, NQ);
    transpose_to_f16<<<(n + TPB - 1) / TPB, TPB, 0, stream>>>(Wo, WoT, NQ, D_);
    int nk = D_ * NKV;
    transpose_to_f16<<<(nk + TPB - 1) / TPB, TPB, 0, stream>>>(Wk, WkT, D_, NKV);
    transpose_to_f16<<<(nk + TPB - 1) / TPB, TPB, 0, stream>>>(Wv, WvT, D_, NKV);
  }
  // 3) QKV projections (WMMA, 128x128 block tiles, async-LDS staged)
  gemm_f16_wmma<<<dim3(NQ / 128, M / 128), TPB, 0, stream>>>(xh, WqT, bq, qf, M, NQ, D_);
  gemm_f16_wmma<<<dim3(NKV / 128, M / 128), TPB, 0, stream>>>(xh, WkT, bk, kf, M, NKV, D_);
  gemm_f16_wmma<<<dim3(NKV / 128, M / 128), TPB, 0, stream>>>(xh, WvT, bv, vf, M, NKV, D_);

  // 4) zero K/V caches (positions >= S stay zero, they're masked anyway)
  hipMemsetAsync(Kh, 0, (size_t)B_ * KVH_ * MAXC * HD_ * 2, stream);
  hipMemsetAsync(Vt, 0, (size_t)B_ * KVH_ * HD_ * MAXC * 2, stream);

  // 5) RoPE + cache scatter
  {
    int nq = B_ * S_ * H_ * HD_;
    rope_q_kernel<<<(nq + TPB - 1) / TPB, TPB, 0, stream>>>(qf, sidx, Qh);
    int nkv = B_ * S_ * KVH_ * HD_;
    rope_k_kernel<<<(nkv + TPB - 1) / TPB, TPB, 0, stream>>>(kf, sidx, Kh);
    v_scatter_kernel<<<(nkv + TPB - 1) / TPB, TPB, 0, stream>>>(vf, sidx, Vt);
  }

  // 6) attention (dynamic LDS: 16x2048 f32 scores + reduction scratch)
  {
    size_t smem = (size_t)(16 * MAXC + 256 + 16 + 16) * sizeof(float);
    hipFuncSetAttribute((const void*)attn_kernel,
                        hipFuncAttributeMaxDynamicSharedMemorySize, (int)smem);
    attn_kernel<<<dim3(S_ / 16, H_, B_), TPB, smem, stream>>>(Qh, Kh, Vt, Oh);
  }

  // 7) output projection
  gemm_f16_wmma<<<dim3(D_ / 128, M / 128), TPB, 0, stream>>>(Oh, WoT, nullptr, out,
                                                             M, D_, NQ);
}
